// LatentMultiScaleODEEncoder_54417235641015
// MI455X (gfx1250) — matrix operand order repaired and checked by reading, not compiled, not run on hardware
//
#include <hip/hip_runtime.h>
#include <hip/hip_bf16.h>
#include <math.h>

typedef __attribute__((ext_vector_type(16))) _Float16 v16h;
typedef __attribute__((ext_vector_type(8)))  _Float16 h8;
typedef __attribute__((ext_vector_type(8)))  float    v8f;
typedef __attribute__((ext_vector_type(4)))  float    f4;
typedef __attribute__((ext_vector_type(4)))  int      i4;

#define BATCH   16
#define SEQT    1024
#define DMODEL  512
#define NH      4
#define DHEAD   128
#define FFDIM   2048
#define NLAT    32
#define NSC     4
#define COUP_C  0.1f

#if defined(__has_builtin)
#if __has_builtin(__builtin_amdgcn_global_load_async_to_lds_b128)
#define HAVE_ASYNC_LDS 1
#endif
#endif

typedef __attribute__((address_space(1))) void* gas_ptr;
typedef __attribute__((address_space(3))) void* las_ptr;
typedef __attribute__((address_space(1))) i4*   gp_i4;   // int4 __device__ *
typedef __attribute__((address_space(3))) i4*   lp_i4;   // int4 __shared__ *

__device__ __forceinline__ void wait_async0() {
#if defined(__has_builtin)
#if __has_builtin(__builtin_amdgcn_s_wait_asynccnt)
  __builtin_amdgcn_s_wait_asynccnt(0);
#else
  asm volatile("s_wait_asynccnt 0x0" ::: "memory");
#endif
#else
  asm volatile("s_wait_asynccnt 0x0" ::: "memory");
#endif
}

__device__ __forceinline__ v8f wmma16(v16h a, v16h b, v8f c) {
  return __builtin_amdgcn_wmma_f32_16x16x32_f16(false, a, false, b, (short)0, c, false, false);
}

__device__ __forceinline__ h8 cvt8(f4 a, f4 b) {
  h8 r;
  r[0] = (_Float16)a[0]; r[1] = (_Float16)a[1]; r[2] = (_Float16)a[2]; r[3] = (_Float16)a[3];
  r[4] = (_Float16)b[0]; r[5] = (_Float16)b[1]; r[6] = (_Float16)b[2]; r[7] = (_Float16)b[3];
  return r;
}

// ---------------------------------------------------------------- utilities
__global__ void cvt_f16_kernel(const float* __restrict__ in, _Float16* __restrict__ out, int n) {
  int i = blockIdx.x * blockDim.x + threadIdx.x;
  if (i < n) out[i] = (_Float16)in[i];
}

__global__ void cvt_f16_t_kernel(const float* __restrict__ in, _Float16* __restrict__ out,
                                 int K, int N, int Kt, int koff) {
  int idx = blockIdx.x * blockDim.x + threadIdx.x;
  if (idx >= K * N) return;
  int k = idx / N, n = idx - k * N;
  out[(size_t)n * Kt + koff + k] = (_Float16)in[idx];
}

__global__ __launch_bounds__(256) void minmax_kernel(const float* __restrict__ t, float* __restrict__ out, int n) {
  __shared__ float smin[8], smax[8];
  float mn = 3.0e38f, mx = -3.0e38f;
  for (int i = threadIdx.x; i < n; i += 256) { float v = t[i]; mn = fminf(mn, v); mx = fmaxf(mx, v); }
  for (int off = 1; off < 32; off <<= 1) { mn = fminf(mn, __shfl_xor(mn, off, 32)); mx = fmaxf(mx, __shfl_xor(mx, off, 32)); }
  if ((threadIdx.x & 31) == 0) { smin[threadIdx.x >> 5] = mn; smax[threadIdx.x >> 5] = mx; }
  __syncthreads();
  if (threadIdx.x == 0) {
    for (int w = 1; w < 8; ++w) { mn = fminf(mn, smin[w]); mx = fmaxf(mx, smax[w]); }
    out[0] = mn; out[1] = mx;
  }
}

__device__ __forceinline__ float te_val(float t, int i) {
  float fr = __expf(-logf(10000.0f) * (float)(i & 15) / 16.0f);
  float a = t * fr;
  return (i < 16) ? sinf(a) : cosf(a);
}

__global__ void embed_pack_kernel(const float* __restrict__ times, const float* __restrict__ values,
                                  const float* __restrict__ mask, float* __restrict__ A, int rows) {
  int idx = blockIdx.x * blockDim.x + threadIdx.x;
  if (idx >= rows * 160) return;
  int row = idx / 160, c = idx - row * 160;
  float v;
  if (c < 64)       v = values[row * 64 + c];
  else if (c < 128) v = mask[row * 64 + (c - 64)];
  else              v = te_val(times[row], c - 128);
  A[idx] = v;
}

__global__ void te_times_kernel(const float* __restrict__ times, float* __restrict__ out, int rows) {
  int idx = blockIdx.x * blockDim.x + threadIdx.x;
  if (idx >= rows * 32) return;
  out[idx] = te_val(times[idx >> 5], idx & 31);
}

__global__ void te_latent_kernel(const float* __restrict__ tmm, float* __restrict__ out) {
  int idx = blockIdx.x * blockDim.x + threadIdx.x;
  if (idx >= NLAT * 32) return;
  int l = idx >> 5, i = idx & 31;
  float t = tmm[0] + (tmm[1] - tmm[0]) * (float)l / (float)(NLAT - 1);
  out[idx] = te_val(t, i);
}

__global__ void qin_add_kernel(const float* __restrict__ zflat, const float* __restrict__ tq,
                               float* __restrict__ qin) {
  int idx = blockIdx.x * blockDim.x + threadIdx.x;
  if (idx >= BATCH * NLAT * DMODEL) return;
  int row = idx >> 9;
  int l = row & (NLAT - 1);
  int c = idx & (DMODEL - 1);
  qin[idx] = zflat[idx] + tq[l * DMODEL + c];
}

__global__ __launch_bounds__(256) void ln_kernel(const float* __restrict__ X, const float* __restrict__ R,
                                                 const float* __restrict__ g, const float* __restrict__ bta,
                                                 float* __restrict__ out, int rows) {
  int wave = threadIdx.x >> 5, lane = threadIdx.x & 31;
  int row = blockIdx.x * 8 + wave;
  if (row >= rows) return;
  const float* xp = X + (size_t)row * DMODEL;
  const float* rp = R ? (R + (size_t)row * DMODEL) : nullptr;
  float v[16]; float s = 0.f, sq = 0.f;
#pragma unroll
  for (int j = 0; j < 16; ++j) {
    float t = xp[lane + j * 32];
    if (rp) t += rp[lane + j * 32];
    v[j] = t; s += t; sq += t * t;
  }
  for (int off = 1; off < 32; off <<= 1) { s += __shfl_xor(s, off, 32); sq += __shfl_xor(sq, off, 32); }
  float mean = s * (1.0f / DMODEL);
  float var  = sq * (1.0f / DMODEL) - mean * mean;
  float rstd = rsqrtf(var + 1e-5f);
  float* op = out + (size_t)row * DMODEL;
#pragma unroll
  for (int j = 0; j < 16; ++j) {
    int c = lane + j * 32;
    op[c] = (v[j] - mean) * rstd * g[c] + bta[c];
  }
}

// ------------------------------------------------- generic WMMA f16 GEMM
// Contracts: Kd % 32 == 0, N % 128 == 0.  B is N-major f16: Bt[n*Kd + k].
// Block tile 128x128; 8 waves in 4x2 grid; wave tile 32x64 (8 WMMA / K-step).
__global__ __launch_bounds__(256) void gemm_kernel(const float* __restrict__ A, int lda,
                                                   const _Float16* __restrict__ Bt,
                                                   const float* __restrict__ bias,
                                                   const float* __restrict__ res,
                                                   float* __restrict__ C,
                                                   int M, int N, int Kd, int act,
                                                   int trC, int ldc) {
  __shared__ __align__(16) _Float16 As[128][40];   // [m][k]
  __shared__ __align__(16) _Float16 Bs[128][40];   // [n][k]
  int tid = threadIdx.x;
  int wave = tid >> 5, lane = tid & 31;
  int lhalf = lane >> 4, lidx = lane & 15;
  int mw = wave & 3, nw = wave >> 2;
  int mBase = blockIdx.y * 128;
  int nBase = blockIdx.x * 128;

  v8f acc[2][4];
  v8f zero = {};
#pragma unroll
  for (int i = 0; i < 2; ++i)
#pragma unroll
    for (int j = 0; j < 4; ++j) acc[i][j] = zero;

  const int arow = tid >> 1, aks = (tid & 1) * 16;
  const bool arv = (mBase + arow) < M;
  const float* aptr = A + (size_t)(arv ? (mBase + arow) : 0) * lda + aks;
  const int bc = tid >> 1, bkh = (tid & 1) * 16;      // N%128==0 -> always valid
  const _Float16* bptr = Bt + (size_t)(nBase + bc) * Kd + bkh;

  for (int k0 = 0; k0 < Kd; k0 += 32) {
    __syncthreads();
#ifdef HAVE_ASYNC_LDS
    // async global->LDS copy of B tile (no VGPR round trip), ASYNCcnt-tracked
    __builtin_amdgcn_global_load_async_to_lds_b128(
        (gp_i4)(gas_ptr)(void*)(bptr + k0),
        (lp_i4)(las_ptr)(void*)&Bs[bc][bkh], 0, 0);
    __builtin_amdgcn_global_load_async_to_lds_b128(
        (gp_i4)(gas_ptr)(void*)(bptr + k0 + 8),
        (lp_i4)(las_ptr)(void*)&Bs[bc][bkh + 8], 0, 0);
#else
    {
      const h8* bp = (const h8*)(bptr + k0);
      h8 b0 = bp[0], b1 = bp[1];
      *(h8*)&Bs[bc][bkh]     = b0;
      *(h8*)&Bs[bc][bkh + 8] = b1;
    }
#endif
    { // stage A: 16 f32 -> 2 h8 stores (f32->f16 convert, VGPR path)
      h8 a0 = {}, a1 = {};
      if (arv) {
        const f4* ap = (const f4*)(aptr + k0);
        a0 = cvt8(ap[0], ap[1]);
        a1 = cvt8(ap[2], ap[3]);
      }
      *(h8*)&As[arow][aks]     = a0;
      *(h8*)&As[arow][aks + 8] = a1;
    }
#ifdef HAVE_ASYNC_LDS
    wait_async0();
#endif
    __syncthreads();

    v16h af0, af1;
    {
      int m  = mw * 32 + lidx;
      int kb = lhalf * 8;
      h8 x0 = *(const h8*)&As[m][kb];
      h8 x1 = *(const h8*)&As[m][kb + 16];
      h8 y0 = *(const h8*)&As[m + 16][kb];
      h8 y1 = *(const h8*)&As[m + 16][kb + 16];
#pragma unroll
      for (int v = 0; v < 8; ++v) {
        af0[v] = x0[v]; af0[v + 8] = x1[v];
        af1[v] = y0[v]; af1[v + 8] = y1[v];
      }
    }
#pragma unroll
    for (int nt = 0; nt < 4; ++nt) {
      v16h bf;
      int c  = nw * 64 + nt * 16 + lidx;
      int kb = lhalf * 8;
      h8 b0 = *(const h8*)&Bs[c][kb];
      h8 b1 = *(const h8*)&Bs[c][kb + 16];
#pragma unroll
      for (int v = 0; v < 8; ++v) { bf[v] = b0[v]; bf[v + 8] = b1[v]; }
      acc[0][nt] = wmma16(af0, bf, acc[0][nt]);
      acc[1][nt] = wmma16(af1, bf, acc[1][nt]);
    }
  }
  // epilogue
#pragma unroll
  for (int mf = 0; mf < 2; ++mf) {
#pragma unroll
    for (int nt = 0; nt < 4; ++nt) {
#pragma unroll
      for (int r = 0; r < 8; ++r) {
        int row = mBase + mw * 32 + mf * 16 + r + lhalf * 8;
        int col = nBase + nw * 64 + nt * 16 + lidx;
        if (row < M && col < N) {
          float v = acc[mf][nt][r];
          if (bias) v += bias[col];
          if (act == 1) {
            float x = v;
            v = 0.5f * x * (1.0f + tanhf(0.7978845608028654f * (x + 0.044715f * x * x * x)));
          } else if (act == 2) {
            v = tanhf(v);
          }
          if (res) v += res[(size_t)row * N + col];
          if (trC) C[(size_t)col * ldc + row] = v;
          else     C[(size_t)row * ldc + col] = v;
        }
      }
    }
  }
}

// ------------------------------------------------- flash attention (WMMA)
// Q,K,O: (B*L, Hh*128) f32 row-major.  V transposed: Vt[(h*128+d)*ldv + b*Lk + key]
// Requires Lk % 32 == 0.
__global__ __launch_bounds__(256) void flash_attn_kernel(const float* __restrict__ Q,
                                                         const float* __restrict__ Km,
                                                         const float* __restrict__ Vt, int ldv,
                                                         float* __restrict__ O,
                                                         int Lq, int Lk, int Hh, float scale) {
  int nqblk = (Lq + 127) / 128;
  int bid  = blockIdx.x;
  int qblk = bid % nqblk;
  int h    = (bid / nqblk) % Hh;
  int b    = bid / (nqblk * Hh);
  int Hdim = Hh * DHEAD;

  int tid = threadIdx.x;
  int wave = tid >> 5, lane = tid & 31;
  int lhalf = lane >> 4, lidx = lane & 15;

  __shared__ __align__(16) _Float16 Ks[32][DHEAD + 8];
  __shared__ __align__(16) _Float16 Vs[DHEAD][40];
  __shared__ __align__(16) _Float16 Ps[8][16][36];

  int q0 = qblk * 128 + wave * 16;

  v16h qf[4];
  {
    int grow = q0 + lidx;
    bool qv = grow < Lq;
    const float* qp = Q + ((size_t)(b * Lq + (qv ? grow : 0))) * Hdim + h * DHEAD;
    int kb = lhalf * 8;
#pragma unroll
    for (int ch = 0; ch < 4; ++ch) {
#pragma unroll
      for (int v = 0; v < 8; ++v) {
        int k = ch * 32 + kb + (v & 3) * 2 + (v >> 2) * 16;
        float x0 = qv ? qp[k] : 0.f;
        float x1 = qv ? qp[k + 1] : 0.f;
        qf[ch][2 * v]     = (_Float16)(x0 * scale);
        qf[ch][2 * v + 1] = (_Float16)(x1 * scale);
      }
    }
  }

  v8f zero = {};
  v8f oacc[8];
#pragma unroll
  for (int i = 0; i < 8; ++i) oacc[i] = zero;
  float mrow[8], lrow[8];
#pragma unroll
  for (int r = 0; r < 8; ++r) { mrow[r] = -3.0e38f; lrow[r] = 0.f; }

  const int skey = tid >> 3, sc0 = (tid & 7) * 16;
  const float* kbase = Km + ((size_t)(b * Lk + skey)) * Hdim + h * DHEAD + sc0;
  const int sd = tid >> 1, skh = (tid & 1) * 16;
  const float* vbase = Vt + (size_t)(h * DHEAD + sd) * ldv + b * Lk + skh;

  for (int kt = 0; kt < Lk; kt += 32) {
    __syncthreads();
    {
      const f4* kp = (const f4*)(kbase + (size_t)kt * Hdim);
      *(h8*)&Ks[skey][sc0]     = cvt8(kp[0], kp[1]);
      *(h8*)&Ks[skey][sc0 + 8] = cvt8(kp[2], kp[3]);
      const f4* vp = (const f4*)(vbase + kt);
      *(h8*)&Vs[sd][skh]     = cvt8(vp[0], vp[1]);
      *(h8*)&Vs[sd][skh + 8] = cvt8(vp[2], vp[3]);
    }
    __syncthreads();

    v8f s[2];
#pragma unroll
    for (int nt = 0; nt < 2; ++nt) {
      v8f sc = zero;
#pragma unroll
      for (int ch = 0; ch < 4; ++ch) {
        v16h bf;
        int keyn = nt * 16 + lidx;
        int kb = ch * 32 + lhalf * 8;
        h8 b0 = *(const h8*)&Ks[keyn][kb];
        h8 b1 = *(const h8*)&Ks[keyn][kb + 16];
#pragma unroll
        for (int v = 0; v < 8; ++v) { bf[v] = b0[v]; bf[v + 8] = b1[v]; }
        sc = wmma16(qf[ch], bf, sc);
      }
      s[nt] = sc;
    }

#pragma unroll
    for (int r = 0; r < 8; ++r) {
      float mx = fmaxf(s[0][r], s[1][r]);
      for (int off = 1; off < 16; off <<= 1) mx = fmaxf(mx, __shfl_xor(mx, off, 32));
      float mn = fmaxf(mrow[r], mx);
      float corr = __expf(mrow[r] - mn);
      float p0 = __expf(s[0][r] - mn);
      float p1 = __expf(s[1][r] - mn);
      float rs = p0 + p1;
      for (int off = 1; off < 16; off <<= 1) rs += __shfl_xor(rs, off, 32);
      lrow[r] = lrow[r] * corr + rs;
      mrow[r] = mn;
#pragma unroll
      for (int ot = 0; ot < 8; ++ot) oacc[ot][r] *= corr;
      s[0][r] = p0; s[1][r] = p1;
    }

#pragma unroll
    for (int nt = 0; nt < 2; ++nt)
#pragma unroll
      for (int r = 0; r < 8; ++r)
        Ps[wave][r + lhalf * 8][nt * 16 + lidx] = (_Float16)s[nt][r];

    v16h pf;
    {
      int kb = lhalf * 8;
      h8 p0 = *(const h8*)&Ps[wave][lidx][kb];
      h8 p1 = *(const h8*)&Ps[wave][lidx][kb + 16];
#pragma unroll
      for (int v = 0; v < 8; ++v) { pf[v] = p0[v]; pf[v + 8] = p1[v]; }
    }
#pragma unroll
    for (int ot = 0; ot < 8; ++ot) {
      v16h bf;
      int d  = ot * 16 + lidx;
      int kb = lhalf * 8;
      h8 b0 = *(const h8*)&Vs[d][kb];
      h8 b1 = *(const h8*)&Vs[d][kb + 16];
#pragma unroll
      for (int v = 0; v < 8; ++v) { bf[v] = b0[v]; bf[v + 8] = b1[v]; }
      oacc[ot] = wmma16(pf, bf, oacc[ot]);
    }
  }

#pragma unroll
  for (int r = 0; r < 8; ++r) {
    int grow = q0 + r + lhalf * 8;
    if (grow < Lq) {
      float inv = 1.0f / lrow[r];
      float* op = O + ((size_t)(b * Lq + grow)) * Hdim + h * DHEAD;
#pragma unroll
      for (int ot = 0; ot < 8; ++ot) op[ot * 16 + lidx] = oacc[ot][r] * inv;
    }
  }
}

// ------------------------------------------------- latent ODE (RK4, persistent)
__global__ __launch_bounds__(512) void ode_kernel(const float* __restrict__ obs,
                                                  const _Float16* __restrict__ W1,
                                                  const float* __restrict__ b1,
                                                  const _Float16* __restrict__ W2,
                                                  const float* __restrict__ b2,
                                                  const float* __restrict__ scale_emb,
                                                  const float* __restrict__ tmm,
                                                  float* __restrict__ zflat) {
  __shared__ float z[512], ztmp[512], kacc[512];
  __shared__ float hid[4][256];
  __shared__ float te[32];
  __shared__ float se[4][16];
  __shared__ float zmean[128];

  const int b = blockIdx.x;
  const int tid = threadIdx.x;
  const float tmin = tmm[0], tmax = tmm[1];
  const float dt = (tmax - tmin) / (float)(NLAT - 1);
  const float hstep = dt * 0.25f;

  if (tid < 512) {
    float v = obs[(size_t)(b * SEQT) * DMODEL + tid];
    z[tid] = v;
    zflat[((size_t)b * NLAT) * DMODEL + tid] = v;
  }
  if (tid < 64) se[tid >> 4][tid & 15] = scale_emb[tid];
  __syncthreads();

  const float ccf[4] = {0.f, 0.5f, 0.5f, 1.f};
  const float wcf[4] = {1.f, 2.f, 2.f, 1.f};
  const float anx[4] = {0.5f, 0.5f, 1.f, 0.f};

  for (int iv = 0; iv < NLAT - 1; ++iv) {
    float t0 = tmin + dt * (float)iv;
    for (int sub = 0; sub < 4; ++sub) {
      float tstart = t0 + hstep * (float)sub;
      ztmp[tid] = z[tid];
      kacc[tid] = 0.f;
      __syncthreads();
      for (int st = 0; st < 4; ++st) {
        if (tid < 32) te[tid] = te_val(tstart + ccf[st] * hstep, tid);
        if (tid >= 64 && tid < 192) {
          int d = tid - 64;
          zmean[d] = 0.25f * (ztmp[d] + ztmp[128 + d] + ztmp[256 + d] + ztmp[384 + d]);
        }
        __syncthreads();
        {
          int n  = tid & 255;
          int sp = (tid >> 8) * 2;
          float a0 = b1[n], a1 = b1[n];
          for (int k = 0; k < 128; ++k) {
            float w = (float)W1[k * 256 + n];
            a0 += ztmp[sp * 128 + k] * w;
            a1 += ztmp[(sp + 1) * 128 + k] * w;
          }
          for (int k = 0; k < 16; ++k) {
            float w = (float)W1[(128 + k) * 256 + n];
            a0 += se[sp][k] * w;
            a1 += se[sp + 1][k] * w;
          }
          for (int k = 0; k < 32; ++k) {
            float w = (float)W1[(144 + k) * 256 + n];
            float t = te[k];
            a0 += t * w;
            a1 += t * w;
          }
          hid[sp][n]     = tanhf(a0);
          hid[sp + 1][n] = tanhf(a1);
        }
        __syncthreads();
        {
          int i = tid;
          int sc = i >> 7, d = i & 127;
          float o = b2[d];
          for (int k = 0; k < 256; ++k) o += hid[sc][k] * (float)W2[k * 128 + d];
          float kv = o + COUP_C * (zmean[d] - ztmp[i]);
          kacc[i] += wcf[st] * kv;
          if (st < 3) ztmp[i] = z[i] + anx[st] * hstep * kv;
          else        z[i]   += (hstep / 6.0f) * kacc[i];
        }
        __syncthreads();
      }
    }
    zflat[((size_t)b * NLAT + iv + 1) * DMODEL + tid] = z[tid];
    __syncthreads();
  }
}

// ================================================================ host glue
extern "C" void kernel_launch(void* const* d_in, const int* in_sizes, int n_in,
                              void* d_out, int out_size, void* d_ws, size_t ws_size,
                              hipStream_t stream) {
  const float* times = (const float*)d_in[0];
  const float* values = (const float*)d_in[1];
  const float* maskp = (const float*)d_in[2];

  const int ROWS = BATCH * SEQT; // 16384

  char* wsb = (char*)d_ws;
  size_t pos = 0;
  auto walloc = [&](size_t bytes) -> void* {
    void* p = wsb + pos;
    pos += (bytes + 255) & ~(size_t)255;
    return p;
  };
  float* X  = (float*)walloc((size_t)ROWS * DMODEL * 4);
  float* BA = (float*)walloc((size_t)ROWS * DMODEL * 4);
  float* BB = (float*)walloc((size_t)4096 * FFDIM * 4);
  float* BC = (float*)walloc((size_t)ROWS * DMODEL * 4);
  float* BD = (float*)walloc((size_t)ROWS * DMODEL * 4);
  float* TEb  = (float*)walloc((size_t)ROWS * 32 * 4);
  float* EA   = (float*)walloc((size_t)ROWS * 160 * 4);
  float* TL   = (float*)walloc((size_t)NLAT * 32 * 4);
  float* TMM  = (float*)walloc(256);
  float* ZFL  = (float*)walloc((size_t)BATCH * NLAT * DMODEL * 4);
  float* TQ   = (float*)walloc((size_t)NLAT * DMODEL * 4);
  float* QIN  = (float*)walloc((size_t)BATCH * NLAT * DMODEL * 4);
  float* QCA  = (float*)walloc((size_t)BATCH * NLAT * DMODEL * 4);
  float* DCA  = (float*)walloc((size_t)BATCH * NLAT * DMODEL * 4);
  float* PCA  = (float*)walloc((size_t)BATCH * NLAT * DMODEL * 4);
  float* ZREF = (float*)walloc((size_t)BATCH * NLAT * DMODEL * 4);
  float* RTMP = (float*)walloc((size_t)BATCH * DMODEL * 4);
  _Float16* w16 = (_Float16*)walloc((size_t)8 * 1024 * 1024 * 2);

  size_t wo = 0;
  auto a16 = [&](size_t n) -> _Float16* { _Float16* p = w16 + wo; wo += n; return p; };
  auto cvtT = [&](const void* src, _Float16* dst, int K, int N, int Kt, int koff) {
    cvt_f16_t_kernel<<<(K * N + 255) / 256, 256, 0, stream>>>((const float*)src, dst, K, N, Kt, koff);
  };
  auto cvt = [&](const void* src, _Float16* dst, int n) {
    cvt_f16_kernel<<<(n + 255) / 256, 256, 0, stream>>>((const float*)src, dst, n);
  };

  _Float16* embW = a16(DMODEL * 160);
  cvtT(d_in[3], embW, 128, DMODEL, 160, 0);
  cvtT(d_in[5], embW, 32, DMODEL, 160, 128);
  struct LayerW { _Float16 *Wq, *Wk, *Wv, *Wo, *W1, *W2; } lw[2];
  for (int l = 0; l < 2; ++l) {
    int bi = 6 + 12 * l;
    lw[l].Wq = a16(DMODEL * DMODEL); cvtT(d_in[bi + 0], lw[l].Wq, DMODEL, DMODEL, DMODEL, 0);
    lw[l].Wk = a16(DMODEL * DMODEL); cvtT(d_in[bi + 1], lw[l].Wk, DMODEL, DMODEL, DMODEL, 0);
    lw[l].Wv = a16(DMODEL * DMODEL); cvtT(d_in[bi + 2], lw[l].Wv, DMODEL, DMODEL, DMODEL, 0);
    lw[l].Wo = a16(DMODEL * DMODEL); cvtT(d_in[bi + 3], lw[l].Wo, DMODEL, DMODEL, DMODEL, 0);
    lw[l].W1 = a16(DMODEL * FFDIM);  cvtT(d_in[bi + 6], lw[l].W1, DMODEL, FFDIM, DMODEL, 0);
    lw[l].W2 = a16(FFDIM * DMODEL);  cvtT(d_in[bi + 8], lw[l].W2, FFDIM, DMODEL, FFDIM, 0);
  }
  _Float16* odeW1 = a16(176 * 256); cvt(d_in[31], odeW1, 176 * 256);
  _Float16* odeW2 = a16(256 * 128); cvt(d_in[33], odeW2, 256 * 128);
  _Float16* cWtq  = a16(DMODEL * 32); cvtT(d_in[35], cWtq, 32, DMODEL, 32, 0);
  _Float16* cWtk  = a16(DMODEL * 32); cvtT(d_in[36], cWtk, 32, DMODEL, 32, 0);
  _Float16* cWq   = a16(DMODEL * DMODEL); cvtT(d_in[37], cWq, DMODEL, DMODEL, DMODEL, 0);
  _Float16* cWk   = a16(DMODEL * DMODEL); cvtT(d_in[38], cWk, DMODEL, DMODEL, DMODEL, 0);
  _Float16* cWv   = a16(DMODEL * DMODEL); cvtT(d_in[39], cWv, DMODEL, DMODEL, DMODEL, 0);
  _Float16* cWo   = a16(DMODEL * DMODEL); cvtT(d_in[40], cWo, DMODEL, DMODEL, DMODEL, 0);
  _Float16* outW  = a16(DMODEL * DMODEL); cvtT(d_in[43], outW, DMODEL, DMODEL, DMODEL, 0);

  auto gemm = [&](const float* A, int lda, const _Float16* Bw, const float* bias,
                  const float* res, float* C, int M, int N, int Kd, int act,
                  int trC = 0, int ldc = -1) {
    if (ldc < 0) ldc = N;
    dim3 g((N + 127) / 128, (M + 127) / 128);
    gemm_kernel<<<g, 256, 0, stream>>>(A, lda, Bw, bias, res, C, M, N, Kd, act, trC, ldc);
  };

  minmax_kernel<<<1, 256, 0, stream>>>(times, TMM, ROWS);
  embed_pack_kernel<<<(ROWS * 160 + 255) / 256, 256, 0, stream>>>(times, values, maskp, EA, ROWS);
  te_times_kernel<<<(ROWS * 32 + 255) / 256, 256, 0, stream>>>(times, TEb, ROWS);
  te_latent_kernel<<<4, 256, 0, stream>>>(TMM, TL);
  gemm(EA, 160, embW, (const float*)d_in[4], nullptr, X, ROWS, DMODEL, 160, 0);

  const float scale = 0.08838834764831845f; // 1/sqrt(128)
  for (int l = 0; l < 2; ++l) {
    int bi = 6 + 12 * l;
    gemm(X, DMODEL, lw[l].Wq, nullptr, nullptr, BA, ROWS, DMODEL, DMODEL, 0);
    gemm(X, DMODEL, lw[l].Wk, nullptr, nullptr, BC, ROWS, DMODEL, DMODEL, 0);
    gemm(X, DMODEL, lw[l].Wv, nullptr, nullptr, BD, ROWS, DMODEL, DMODEL, 0, 1, ROWS); // Vt
    flash_attn_kernel<<<BATCH * NH * (SEQT / 128), 256, 0, stream>>>(BA, BC, BD, ROWS, BB, SEQT, SEQT, NH, scale);
    gemm(BB, DMODEL, lw[l].Wo, nullptr, nullptr, BA, ROWS, DMODEL, DMODEL, 0);
    ln_kernel<<<ROWS / 8, 256, 0, stream>>>(X, BA, (const float*)d_in[bi + 4], (const float*)d_in[bi + 5], X, ROWS);
    for (int mc = 0; mc < 4; ++mc) {
      const float* xA = X + (size_t)mc * 4096 * DMODEL;
      gemm(xA, DMODEL, lw[l].W1, (const float*)d_in[bi + 7], nullptr, BB, 4096, FFDIM, DMODEL, 1);
      gemm(BB, FFDIM, lw[l].W2, (const float*)d_in[bi + 9], nullptr, BD + (size_t)mc * 4096 * DMODEL, 4096, DMODEL, FFDIM, 0);
    }
    ln_kernel<<<ROWS / 8, 256, 0, stream>>>(X, BD, (const float*)d_in[bi + 10], (const float*)d_in[bi + 11], X, ROWS);
  }

  ode_kernel<<<BATCH, 512, 0, stream>>>(X, odeW1, (const float*)d_in[32], odeW2,
                                        (const float*)d_in[34], (const float*)d_in[30], TMM, ZFL);

  gemm(TEb, 32, cWtk, nullptr, X, BA, ROWS, DMODEL, 32, 0);
  gemm(TL, 32, cWtq, nullptr, nullptr, TQ, NLAT, DMODEL, 32, 0);
  qin_add_kernel<<<(BATCH * NLAT * DMODEL + 255) / 256, 256, 0, stream>>>(ZFL, TQ, QIN);
  gemm(QIN, DMODEL, cWq, nullptr, nullptr, QCA, BATCH * NLAT, DMODEL, DMODEL, 0);
  gemm(BA, DMODEL, cWk, nullptr, nullptr, BC, ROWS, DMODEL, DMODEL, 0);
  gemm(X, DMODEL, cWv, nullptr, nullptr, BD, ROWS, DMODEL, DMODEL, 0, 1, ROWS);
  flash_attn_kernel<<<BATCH * NSC * 1, 256, 0, stream>>>(QCA, BC, BD, ROWS, DCA, NLAT, SEQT, NSC, scale);
  gemm(DCA, DMODEL, cWo, nullptr, nullptr, PCA, BATCH * NLAT, DMODEL, DMODEL, 0);
  ln_kernel<<<(BATCH * NLAT) / 8, 256, 0, stream>>>(ZFL, PCA, (const float*)d_in[41], (const float*)d_in[42], ZREF, BATCH * NLAT);

  gemm(ZREF + (size_t)(NLAT - 1) * DMODEL, NLAT * DMODEL, outW, (const float*)d_in[44],
       nullptr, RTMP, BATCH, DMODEL, DMODEL, 0);
  ln_kernel<<<2, 256, 0, stream>>>(RTMP, nullptr, (const float*)d_in[45], (const float*)d_in[46],
                                   (float*)d_out, BATCH);
}
